// LinearWithLoRA_8555574854313
// MI455X (gfx1250) — matrix operand order repaired and checked by reading, not compiled, not run on hardware
//
#include <hip/hip_runtime.h>
#include <hip/hip_bf16.h>

// ---------------------------------------------------------------------------
// LinearWithLoRA for MI455X (gfx1250, wave32, WMMA + TDM).
//   out[m,n] = sum_k x[m,k]*W[n,k] + b[n] + 2.0 * sum_r low[m,r]*loraB[n,r]
// Compute-bound GEMM (AI ~477 FLOP/B; 144MB working set fits in 192MB L2).
// Precision: bf16 hi/lo split -> 3x v_wmma_f32_16x16x32_bf16 per K=32 step
// (~2^-16 rel err, fp32 accumulate) at bf16 matrix throughput.
// Data path: one-time fp32->bf16(hi,lo) plane conversion into d_ws, then the
// GEMM mainloop uses the Tensor Data Mover (tensor_load_to_lds, 6-arg form)
// to DMA tiles into LDS (double-buffered, wave0-issued, s_wait_tensorcnt +
// barrier), with TDM LDS padding producing a 17-dword row stride.
// Fallback (small ws): inline-convert staging path (round-1 style).
// ---------------------------------------------------------------------------

typedef __attribute__((ext_vector_type(16))) __bf16 v16bf;
typedef __attribute__((ext_vector_type(8)))  float  v8f;
typedef __attribute__((ext_vector_type(4)))  unsigned int v4u;
typedef __attribute__((ext_vector_type(8)))  int    v8i;
typedef __attribute__((ext_vector_type(4)))  int    v4i;
typedef unsigned int u32;
typedef unsigned long long u64;

#define K_DIM   2048
#define N_DIM   2048
#define R_DIM   16
#define SCALING_F 2.0f

#define WGM     128            // workgroup M tile
#define WGN     64             // workgroup N tile
#define KSTEP   32             // K per mainloop iteration (one bf16 WMMA K)
#define LDSROW  17             // dwords per LDS tile row: 16 data + 1 pad
#define XPLANE  (WGM * LDSROW) // 2176 dwords
#define WPLANE  (WGN * LDSROW) // 1088 dwords
#define BUFSZ   (2 * XPLANE + 2 * WPLANE) // 6528 dwords per buffer

__device__ __forceinline__ unsigned short f32_to_bf16_rne(float f) {
    unsigned int u = __float_as_uint(f);
    u += 0x7FFFu + ((u >> 16) & 1u);
    return (unsigned short)(u >> 16);
}
__device__ __forceinline__ float bf16_bits_to_f32(unsigned short h) {
    return __uint_as_float(((unsigned int)h) << 16);
}
__device__ __forceinline__ void split2(float f0, float f1,
                                       unsigned int& hi, unsigned int& lo) {
    unsigned short h0 = f32_to_bf16_rne(f0);
    unsigned short h1 = f32_to_bf16_rne(f1);
    unsigned short l0 = f32_to_bf16_rne(f0 - bf16_bits_to_f32(h0));
    unsigned short l1 = f32_to_bf16_rne(f1 - bf16_bits_to_f32(h1));
    hi = (unsigned int)h0 | ((unsigned int)h1 << 16);
    lo = (unsigned int)l0 | ((unsigned int)l1 << 16);
}

// ---------------------------------------------------------------------------
// Prologue 1: fp32 -> bf16 hi/lo planes (one-time, bandwidth-trivial)
// ---------------------------------------------------------------------------
__global__ __launch_bounds__(256) void split_plane_kernel(
    const float* __restrict__ src, unsigned short* __restrict__ hi,
    unsigned short* __restrict__ lo, int n8)
{
    int t = blockIdx.x * 256 + threadIdx.x;
    if (t >= n8) return;
    const float4* p = (const float4*)src + (size_t)t * 2;
    float4 f0 = p[0], f1 = p[1];
    u32 h0, h1, h2, h3, l0, l1, l2, l3;
    split2(f0.x, f0.y, h0, l0);
    split2(f0.z, f0.w, h1, l1);
    split2(f1.x, f1.y, h2, l2);
    split2(f1.z, f1.w, h3, l3);
    ((uint4*)hi)[t] = make_uint4(h0, h1, h2, h3);
    ((uint4*)lo)[t] = make_uint4(l0, l1, l2, l3);
}

// ---------------------------------------------------------------------------
// Prologue 2: low[m,r] = dot(x[m,:], loraA[r,:])  -- fp32
// ---------------------------------------------------------------------------
__global__ __launch_bounds__(256) void lora_low_kernel(
    const float* __restrict__ x, const float* __restrict__ loraA,
    float* __restrict__ low, int M)
{
    int idx = blockIdx.x * 256 + threadIdx.x;           // idx = m*16 + r
    if (idx >= M * R_DIM) return;
    int m = idx >> 4;
    int r = idx & 15;
    const float4* xp = (const float4*)(x + (size_t)m * K_DIM);
    const float4* ap = (const float4*)(loraA + (size_t)r * K_DIM);
    float s = 0.0f;
    for (int k = 0; k < K_DIM / 4; ++k) {
        float4 a = xp[k], b = ap[k];
        s += a.x * b.x + a.y * b.y + a.z * b.z + a.w * b.w;
    }
    low[idx] = s;
}

// ---------------------------------------------------------------------------
// TDM: issue a 2D bf16 tile load (tile_rows x 32 elems) global -> LDS with
// 1-dword padding after every 16 dwords (row stride becomes 17 dwords).
// ---------------------------------------------------------------------------
__device__ __forceinline__ void tdm_load_tile(
    const u32* ldsDst, const unsigned short* gsrc,
    unsigned tileRows, unsigned tensorRows)
{
    u64 ga = (u64)(size_t)gsrc;
    u32 lds_addr = (u32)(size_t)ldsDst;     // low 32 bits = wave-relative LDS addr

    v4u g0;
    g0[0] = 1u;                                   // count=1 valid, user desc
    g0[1] = lds_addr;                             // LDS byte address
    g0[2] = (u32)(ga & 0xFFFFFFFFu);              // global_addr[31:0]
    g0[3] = (u32)((ga >> 32) & 0x1FFFFFFu)        // global_addr[56:32]
          | 0x80000000u;                          // type = 2 ("image")

    const u32 td0 = K_DIM;                        // tensor_dim0 (elements)
    const u32 td1 = tensorRows;                   // tensor_dim1
    const u64 st0 = K_DIM;                        // tensor_dim0_stride (elements)
    v8i g1;
    g1[0] = (int)((1u << 16)        // data_size = 1 -> 2 bytes
                | (1u << 20)        // pad_enable
                | (3u << 22)        // pad_interval code 3 = every 16 dwords
                | (0u << 25));      // pad_amount  code 0 = 1 dword
    g1[1] = (int)((td0 & 0xFFFFu) << 16);                       // dim0[15:0]
    g1[2] = (int)(((td0 >> 16) & 0xFFFFu) | ((td1 & 0xFFFFu) << 16));
    g1[3] = (int)(((td1 >> 16) & 0xFFFFu) | ((KSTEP & 0xFFFFu) << 16)); // tile_dim0=32
    g1[4] = (int)(tileRows & 0xFFFFu);                          // tile_dim1; tile_dim2=0
    g1[5] = (int)(st0 & 0xFFFFFFFFu);
    g1[6] = (int)((st0 >> 32) & 0xFFFFu);                       // stride0 hi; stride1 lo16=0
    g1[7] = 0;

    v4i g2 = {0, 0, 0, 0};
    v4i g3 = {0, 0, 0, 0};
    v8i g4 = {0, 0, 0, 0, 0, 0, 0, 0};
    // 6-arg form (clang-23 / therock-10.0 headers): extra int32x8 group + cpol
    __builtin_amdgcn_tensor_load_to_lds(g0, g1, g2, g3, g4, 0);
}

__device__ __forceinline__ void issue_tile_loads(
    const u32* buf,
    const unsigned short* xhi, const unsigned short* xlo,
    const unsigned short* whi, const unsigned short* wlo,
    int mBase, int nBase, int k0, int M)
{
    tdm_load_tile(buf,                        xhi + (size_t)mBase * K_DIM + k0, WGM, M);
    tdm_load_tile(buf + XPLANE,               xlo + (size_t)mBase * K_DIM + k0, WGM, M);
    tdm_load_tile(buf + 2 * XPLANE,           whi + (size_t)nBase * K_DIM + k0, WGN, N_DIM);
    tdm_load_tile(buf + 2 * XPLANE + WPLANE,  wlo + (size_t)nBase * K_DIM + k0, WGN, N_DIM);
}

// ---------------------------------------------------------------------------
// One K=32 step: gather operands from LDS, 4 C tiles x 3 split-products.
// ---------------------------------------------------------------------------
__device__ __forceinline__ void gemm_step(
    const u32* buf, int waveM, int waveN, int laneHalf, int l15, v8f acc[2][2])
{
    const u32* sXhi = buf;
    const u32* sXlo = buf + XPLANE;
    const u32* sWhi = buf + 2 * XPLANE;
    const u32* sWlo = buf + 2 * XPLANE + WPLANE;

    union U { v16bf v; unsigned int u[8]; };
    U aHi[2], aLo[2], bHi[2], bLo[2];
#pragma unroll
    for (int im = 0; im < 2; ++im) {
        // 16-bit A-matrix 16x32: VGPR v, lanes0-15 K-pairs {0-3,8-11},
        // lanes16-31 K-pairs {4-7,12-15}
        int row = waveM + im * 16 + l15;
        int rb  = row * LDSROW + laneHalf * 4;
#pragma unroll
        for (int v = 0; v < 8; ++v) {
            int pair = (v & 3) + ((v & 4) << 1);  // v<4 ? v : v+4
            aHi[im].u[v] = sXhi[rb + pair];
            aLo[im].u[v] = sXlo[rb + pair];
        }
    }
#pragma unroll
    for (int in_ = 0; in_ < 2; ++in_) {
        // 16-bit B-matrix 32x16: lanes0-15 K-pairs {0-3,8-11},
        // lanes16-31 K-pairs {4-7,12-15} (same as A).  B[k][n] = W[n][k].
        int row = waveN + in_ * 16 + l15;
        int rb  = row * LDSROW + laneHalf * 4;
#pragma unroll
        for (int v = 0; v < 8; ++v) {
            bHi[in_].u[v] = sWhi[rb + (v & 3) + ((v & 4) << 1)];
            bLo[in_].u[v] = sWlo[rb + (v & 3) + ((v & 4) << 1)];
        }
    }
#pragma unroll
    for (int im = 0; im < 2; ++im) {
#pragma unroll
        for (int in_ = 0; in_ < 2; ++in_) {
            acc[im][in_] = __builtin_amdgcn_wmma_f32_16x16x32_bf16(
                false, aHi[im].v, false, bHi[in_].v, (short)0,
                acc[im][in_], false, false);
            acc[im][in_] = __builtin_amdgcn_wmma_f32_16x16x32_bf16(
                false, aHi[im].v, false, bLo[in_].v, (short)0,
                acc[im][in_], false, false);
            acc[im][in_] = __builtin_amdgcn_wmma_f32_16x16x32_bf16(
                false, aLo[im].v, false, bHi[in_].v, (short)0,
                acc[im][in_], false, false);
        }
    }
}

// ---------------------------------------------------------------------------
// Main fused GEMM.  grid = (N/WGN, M/WGM), block = 256 (8 waves, 4Mx2N).
// ---------------------------------------------------------------------------
template <bool USE_TDM>
__global__ __launch_bounds__(256) void lora_gemm_kernel(
    const float* __restrict__ x, const float* __restrict__ W,
    const float* __restrict__ bias, const float* __restrict__ loraB,
    const float* __restrict__ low,
    const unsigned short* __restrict__ xhi, const unsigned short* __restrict__ xlo,
    const unsigned short* __restrict__ whi, const unsigned short* __restrict__ wlo,
    float* __restrict__ out, int M)
{
    __shared__ u32 lds[2 * BUFSZ];    // 52,224 B (double buffer for TDM path)

    const int tid      = threadIdx.x;
    const int lane     = tid & 31;
    const int wave     = tid >> 5;
    const int laneHalf = lane >> 4;
    const int l15      = lane & 15;

    const int mBase = blockIdx.y * WGM;
    const int nBase = blockIdx.x * WGN;
    const int waveM = (wave >> 1) * 32;
    const int waveN = (wave & 1) * 32;

    v8f acc[2][2] = {};

    if constexpr (USE_TDM) {
        constexpr int NSTEPS = K_DIM / KSTEP;
        if (wave == 0) {
            issue_tile_loads(lds, xhi, xlo, whi, wlo, mBase, nBase, 0, M);
            __builtin_amdgcn_s_wait_tensorcnt(0);
        }
        __syncthreads();
        int cur = 0;
        for (int s = 0; s < NSTEPS; ++s) {
            const u32* buf = lds + cur * BUFSZ;
            const bool more = (s + 1 < NSTEPS);
            if (wave == 0 && more)
                issue_tile_loads(lds + (cur ^ 1) * BUFSZ, xhi, xlo, whi, wlo,
                                 mBase, nBase, (s + 1) * KSTEP, M);
            gemm_step(buf, waveM, waveN, laneHalf, l15, acc);
            if (wave == 0 && more)
                __builtin_amdgcn_s_wait_tensorcnt(0);
            __syncthreads();
            cur ^= 1;
        }
    } else {
        // fallback: inline fp32->bf16 conversion staging, single buffer
        const int xRow    = tid >> 1;
        const int xHalfK  = (tid & 1) * 16;
        const int wRow    = tid >> 2;
        const int wQuartK = (tid & 3) * 8;
        const float* xg = x + (size_t)(mBase + xRow) * K_DIM + xHalfK;
        const float* wg = W + (size_t)(nBase + wRow) * K_DIM + wQuartK;
        u32* sXhi = lds;
        u32* sXlo = lds + XPLANE;
        u32* sWhi = lds + 2 * XPLANE;
        u32* sWlo = lds + 2 * XPLANE + WPLANE;

        for (int k0 = 0; k0 < K_DIM; k0 += KSTEP) {
            {
                const float4* p = (const float4*)(xg + k0);
                float4 f0 = p[0], f1 = p[1], f2 = p[2], f3 = p[3];
                u32 hi[8], lo[8];
                split2(f0.x, f0.y, hi[0], lo[0]);
                split2(f0.z, f0.w, hi[1], lo[1]);
                split2(f1.x, f1.y, hi[2], lo[2]);
                split2(f1.z, f1.w, hi[3], lo[3]);
                split2(f2.x, f2.y, hi[4], lo[4]);
                split2(f2.z, f2.w, hi[5], lo[5]);
                split2(f3.x, f3.y, hi[6], lo[6]);
                split2(f3.z, f3.w, hi[7], lo[7]);
                int base = xRow * LDSROW + (xHalfK >> 1);
#pragma unroll
                for (int j = 0; j < 8; ++j) { sXhi[base + j] = hi[j]; sXlo[base + j] = lo[j]; }
            }
            {
                const float4* p = (const float4*)(wg + k0);
                float4 f0 = p[0], f1 = p[1];
                u32 hi[4], lo[4];
                split2(f0.x, f0.y, hi[0], lo[0]);
                split2(f0.z, f0.w, hi[1], lo[1]);
                split2(f1.x, f1.y, hi[2], lo[2]);
                split2(f1.z, f1.w, hi[3], lo[3]);
                int base = wRow * LDSROW + (wQuartK >> 1);
#pragma unroll
                for (int j = 0; j < 4; ++j) { sWhi[base + j] = hi[j]; sWlo[base + j] = lo[j]; }
            }
            if (k0 + KSTEP < K_DIM) {
                __builtin_prefetch(xg + k0 + KSTEP, 0, 0);
                __builtin_prefetch(wg + k0 + KSTEP, 0, 0);
            }
            __syncthreads();
            gemm_step(lds, waveM, waveN, laneHalf, l15, acc);
            __syncthreads();
        }
    }

    // ---- epilogue: + bias + SCALING * low @ loraB^T, fused store ----
    // C/D layout: VGPR i -> M = i (lanes 0-15) or 8+i (lanes 16-31), N = lane&15
#pragma unroll
    for (int in_ = 0; in_ < 2; ++in_) {
        const int col = nBase + waveN + in_ * 16 + l15;
        const float bv = bias[col];
        const float4* lbp = (const float4*)(loraB + (size_t)col * R_DIM);
        const float4 lb0 = lbp[0], lb1 = lbp[1], lb2 = lbp[2], lb3 = lbp[3];
#pragma unroll
        for (int im = 0; im < 2; ++im) {
#pragma unroll
            for (int i = 0; i < 8; ++i) {
                const int row = mBase + waveM + im * 16 + laneHalf * 8 + i;
                const float4* lwp = (const float4*)(low + (size_t)row * R_DIM);
                const float4 l0 = lwp[0], l1 = lwp[1], l2 = lwp[2], l3 = lwp[3];
                float s = l0.x * lb0.x + l0.y * lb0.y + l0.z * lb0.z + l0.w * lb0.w
                        + l1.x * lb1.x + l1.y * lb1.y + l1.z * lb1.z + l1.w * lb1.w
                        + l2.x * lb2.x + l2.y * lb2.y + l2.z * lb2.z + l2.w * lb2.w
                        + l3.x * lb3.x + l3.y * lb3.y + l3.z * lb3.z + l3.w * lb3.w;
                out[(size_t)row * N_DIM + col] = acc[im][in_][i] + bv + SCALING_F * s;
            }
        }
    }
}

extern "C" void kernel_launch(void* const* d_in, const int* in_sizes, int n_in,
                              void* d_out, int out_size, void* d_ws, size_t ws_size,
                              hipStream_t stream) {
    const float* x     = (const float*)d_in[0];
    const float* W     = (const float*)d_in[1];
    const float* b     = (const float*)d_in[2];
    const float* loraA = (const float*)d_in[3];
    const float* loraB = (const float*)d_in[4];
    float* out = (float*)d_out;

    const int M = in_sizes[0] / K_DIM;               // 8192

    char* wsb = (char*)d_ws;
    float* low = (float*)wsb;
    const size_t lowBytes = (size_t)M * R_DIM * sizeof(float);        // 512 KB
    unsigned short* xhi = (unsigned short*)(wsb + lowBytes);
    unsigned short* xlo = xhi + (size_t)M * K_DIM;
    unsigned short* whi = xlo + (size_t)M * K_DIM;
    unsigned short* wlo = whi + (size_t)N_DIM * K_DIM;
    const size_t need = lowBytes + 2 * (size_t)M * K_DIM * 2
                                 + 2 * (size_t)N_DIM * K_DIM * 2;     // ~80.5 MB

    lora_low_kernel<<<(M * R_DIM + 255) / 256, 256, 0, stream>>>(x, loraA, low, M);

    dim3 grid(N_DIM / WGN, M / WGM);                 // (32, 64)
    if (ws_size >= need) {
        const int nx8 = (M * K_DIM) / 8;
        const int nw8 = (N_DIM * K_DIM) / 8;
        split_plane_kernel<<<(nx8 + 255) / 256, 256, 0, stream>>>(x, xhi, xlo, nx8);
        split_plane_kernel<<<(nw8 + 255) / 256, 256, 0, stream>>>(W, whi, wlo, nw8);
        lora_gemm_kernel<true><<<grid, 256, 0, stream>>>(
            x, W, b, loraB, low, xhi, xlo, whi, wlo, out, M);
    } else {
        lora_gemm_kernel<false><<<grid, 256, 0, stream>>>(
            x, W, b, loraB, low, xhi, xlo, whi, wlo, out, M);
    }
}
